// LTConvLayer_78743930405432
// MI455X (gfx1250) — compile-verified, matrix operand order checked by reading, not run on hardware
//
#include <hip/hip_runtime.h>

typedef __attribute__((ext_vector_type(16))) _Float16 v16h;
typedef __attribute__((ext_vector_type(8)))  float    v8f;

#define BB    8
#define CIN   64
#define HH    32
#define WW    32
#define OC    64
#define QQ    16
#define LEAF  144
#define LPAD  256
#define MTOT  (BB*HH*WW)     // 8192 pixels total
#define MT    64             // pixels per workgroup (2 image rows)
#define NWAVE (MT/16)        // 4 waves, one 16-pixel WMMA tile each
#define BLOCK (NWAVE*32)     // 128 threads
#define APAD  264            // A-tile row stride in f16 (132 dwords, +4 bank shift/row)
#define HPAD  65             // h0 row stride in f32 (+1 bank shift/row)

// A tile (64 x 256 f16, padded) and the layer-0 output (64 x 64 f32) are
// live in disjoint phases -> overlay them (barrier-protected).
struct alignas(16) SMem {
  union {
    _Float16 A[MT][APAD];    // 33792 B
    float    Hh[MT][HPAD];   // 16640 B
  };
};

__device__ __forceinline__ float fast_tanh(float x) {
#if __has_builtin(__builtin_amdgcn_tanhf)
  return __builtin_amdgcn_tanhf(x);               // 1 TRANS op
#else
  // tanh(x) = (1 - e^-2x) / (1 + e^-2x): v_exp_f32 + v_rcp_f32 (2 TRANS ops)
  float e = __builtin_amdgcn_exp2f(x * -2.8853900817779268f); // exp(-2x)
  return (1.0f - e) * __builtin_amdgcn_rcpf(1.0f + e);
#endif
}

__global__ __launch_bounds__(BLOCK)
void lutconv_wmma_kernel(const float* __restrict__ x,
                         const int*   __restrict__ cidx,
                         const float* __restrict__ w0, const float* __restrict__ b0,
                         const float* __restrict__ w1, const float* __restrict__ b1,
                         const float* __restrict__ w2, const float* __restrict__ b2,
                         const float* __restrict__ w3, const float* __restrict__ b3,
                         float* __restrict__ out)
{
  __shared__ SMem sm;
  __shared__ int  sci[QQ];

  const int t    = threadIdx.x;
  const int oc   = blockIdx.y;
  const int m0   = blockIdx.x * MT;     // first global pixel of this tile
  const int bimg = m0 >> 10;            // image (batch) index: 1024 px per image
  const int y0   = (m0 >> 5) & 31;      // first of the 2 rows in this tile

  if (t < QQ) sci[t] = cidx[oc * QQ + t];
  __syncthreads();

  // ---------------- Phase 1: stage gathered patch tile into LDS (f16) -------
  // Element (p, k): p = pixel in tile, k = (q*9 + kh*3 + kw) for k < 144 else 0.
  // Threads keep p fixed and sweep k, so for a fixed k the 64 lanes read
  // consecutive x-columns -> coalesced 32-wide rows from global (L2-resident).
  {
    const int p    = t & (MT - 1);
    const int yrow = y0 + (p >> 5);
    const int xcol = p & 31;
    const int kk0  = t >> 6;            // 0 or 1
    const float* xb = x + (size_t)bimg * CIN * HH * WW;
#pragma unroll 4
    for (int i = 0; i < LPAD / 2; ++i) {
      const int k = 2 * i + kk0;
      float v = 0.0f;
      if (k < LEAF) {
        const int q  = k / 9;
        const int r  = k - 9 * q;
        const int kh = r / 3;
        const int kw = r - 3 * kh;
        const int yy = yrow + kh - 1;
        const int xx = xcol + kw - 1;
        if ((unsigned)yy < HH && (unsigned)xx < WW)
          v = xb[((size_t)sci[q] * HH + yy) * WW + xx];
      }
      sm.A[p][k] = (_Float16)v;
    }
  }
  __syncthreads();

  // ---------------- Phase 2: layer 0 via v_wmma_f32_16x16x32_f16 ------------
  // h0(M x 64) = flat(M x 256) @ blockdiag(w0)(256 x 64) + b0.
  // N-tile j (groups 16j..16j+15) only touches K in [64j, 64j+64) -> 2 K-steps.
  const int wv    = t >> 5;             // wave id -> pixel tile 16*wv .. +15
  const int lane  = t & 31;
  const int nrel  = lane & 15;
  const int khalf = lane >> 4;

  union AF { v16h h; uint4 u[2]; };
  union BF { v16h h; _Float16 e[16]; };

  const float* w0oc = w0 + (size_t)oc * 64 * 4;
  const float* b0oc = b0 + (size_t)oc * 64;

  v8f acc[4];
#pragma unroll
  for (int j = 0; j < 4; ++j) {
    // C/D layout: lane = column (group 16j+nrel), 8 row-VGPRs -> broadcast bias.
    const float bias = b0oc[16 * j + nrel];
#pragma unroll
    for (int r = 0; r < 8; ++r) acc[j][r] = bias;

#pragma unroll
    for (int s = 0; s < 2; ++s) {
      const int kb = 64 * j + 32 * s;

      // A fragment per ISA 16-bit A layout: lane m = nrel (pixel 16*wv+nrel),
      // lanes 0-15 hold K kb+0..7 & kb+16..23, lanes 16-31 hold kb+8..15 & kb+24..31.
      const int p = 16 * wv + nrel;
      AF a;
      a.u[0] = *(const uint4*)&sm.A[p][kb + 8 * khalf];
      a.u[1] = *(const uint4*)&sm.A[p][kb + 16 + 8 * khalf];

      // B fragment: block-diagonal weights. Column n = 16j+nrel has its 4
      // nonzeros at k_local = 4*nrel - 32*s .. +3 (one 8-aligned slot group,
      // owned by exactly one khalf).
      BF bf;
#pragma unroll
      for (int e = 0; e < 16; ++e) bf.e[e] = (_Float16)0.0f;
      const int knz = 4 * nrel - 32 * s;
      if (knz >= 0 && knz < 32) {
        const int base = 8 * khalf;
#pragma unroll
        for (int c = 0; c < 4; ++c) {
          const int k = knz + c;
          const float wv0 = w0oc[(16 * j + nrel) * 4 + c];
          if (k >= base && k < base + 8)
            bf.e[k - base] = (_Float16)wv0;
          else if (k >= 16 + base && k < 24 + base)
            bf.e[8 + (k - 16 - base)] = (_Float16)wv0;
        }
      }

      acc[j] = __builtin_amdgcn_wmma_f32_16x16x32_f16(
          false, a.h, false, bf.h, (short)0, acc[j], false, false);
    }
  }

  __syncthreads();   // A tile dead -> safe to overlay with Hh

  // ---------------- Phase 3: tanh + transpose h0 through LDS ----------------
#pragma unroll
  for (int j = 0; j < 4; ++j) {
#pragma unroll
    for (int r = 0; r < 8; ++r) {
      const int p = 16 * wv + r + 8 * khalf;   // C layout row -> pixel
      sm.Hh[p][16 * j + nrel] = fast_tanh(acc[j][r]);
    }
  }
  __syncthreads();

  // ---------------- Phase 4: layers 1..3 (16 -> 4 -> 1), 1 thread/pixel -----
  if (t < MT) {
    const int p = t;
    const float* w1oc = w1 + (size_t)oc * 16 * 4;
    const float* b1oc = b1 + (size_t)oc * 16;
    float h1[16];
#pragma unroll
    for (int g = 0; g < 16; ++g) {
      float a = b1oc[g];
#pragma unroll
      for (int n = 0; n < 4; ++n) a += sm.Hh[p][4 * g + n] * w1oc[4 * g + n];
      h1[g] = fast_tanh(a);
    }

    const float* w2oc = w2 + (size_t)oc * 4 * 4;
    const float* b2oc = b2 + (size_t)oc * 4;
    float h2[4];
#pragma unroll
    for (int g = 0; g < 4; ++g) {
      float a = b2oc[g];
#pragma unroll
      for (int n = 0; n < 4; ++n) a += h1[4 * g + n] * w2oc[4 * g + n];
      h2[g] = fast_tanh(a);
    }

    const float* w3oc = w3 + (size_t)oc * 4;
    float a = b3[oc];
#pragma unroll
    for (int n = 0; n < 4; ++n) a += h2[n] * w3oc[n];
    const float res = fast_tanh(a);

    // out is (B, OC, H, W)
    const int m    = m0 + p;
    const int y    = (m >> 5) & 31;
    const int xcol = m & 31;
    out[(((size_t)bimg * OC + oc) * HH + y) * WW + xcol] = res;
  }
}

extern "C" void kernel_launch(void* const* d_in, const int* in_sizes, int n_in,
                              void* d_out, int out_size, void* d_ws, size_t ws_size,
                              hipStream_t stream)
{
  (void)in_sizes; (void)n_in; (void)out_size; (void)d_ws; (void)ws_size;
  const float* x  = (const float*)d_in[0];
  const int*   ci = (const int*)  d_in[1];
  const float* w0 = (const float*)d_in[2];
  const float* b0 = (const float*)d_in[3];
  const float* w1 = (const float*)d_in[4];
  const float* b1 = (const float*)d_in[5];
  const float* w2 = (const float*)d_in[6];
  const float* b2 = (const float*)d_in[7];
  const float* w3 = (const float*)d_in[8];
  const float* b3 = (const float*)d_in[9];
  float* out = (float*)d_out;

  dim3 grid(MTOT / MT, OC);   // 128 pixel-tiles x 64 output channels
  dim3 block(BLOCK);          // 128 threads = 4 wave32
  hipLaunchKernelGGL(lutconv_wmma_kernel, grid, block, 0, stream,
                     x, ci, w0, b0, w1, b1, w2, b2, w3, b3, out);
}
